// EnhancedPatchMoE_71665824301683
// MI455X (gfx1250) — matrix operand order repaired and checked by reading, not compiled, not run on hardware
//
#include <hip/hip_runtime.h>

// ============================================================================
// EnhancedPatchMoE forward for MI455X (gfx1250, wave32, WMMA + TDM).
// bf16 WMMA GEMM, f32 accumulate, double-buffered LDS filled by the
// Tensor Data Mover (tensor_load_to_lds), bf16-preconverted operands.
// ============================================================================

typedef __bf16 bf16;
typedef __attribute__((ext_vector_type(4)))  bf16  bf16x4;
typedef __attribute__((ext_vector_type(8)))  bf16  bf16x8;
typedef __attribute__((ext_vector_type(16))) bf16  bf16x16;
typedef __attribute__((ext_vector_type(8)))  float f32x8;
typedef __attribute__((ext_vector_type(4)))  unsigned int u32x4;
typedef __attribute__((ext_vector_type(8)))  int i32x8;
typedef __attribute__((ext_vector_type(4)))  int i32x4;

#define B_     64
#define L_     100
#define NTOK   6400
#define D_     768
#define D3_    2304
#define D4_    3072
#define NH_    12
#define DH_    64
#define NE_    8
#define HID_   1536
#define DHALF_ 384

#define TM 256           // block tile M (8 waves as 4x2, wave tile 64x64)
#define TN 128           // block tile N
#define TK 32            // K step (bf16)
#define LDSS 40          // padded LDS row stride in bf16 (80 B, 16B aligned)

// ---------------------------------------------------------------------------
// fp32 -> bf16 elementwise (n multiple of 4 in all uses)
// ---------------------------------------------------------------------------
__global__ __launch_bounds__(256)
void f32_to_bf16(const float* __restrict__ in, bf16* __restrict__ out, long n)
{
    long i = ((long)blockIdx.x * 256 + threadIdx.x) * 4;
    if (i + 3 < n) {
        float4 v = *(const float4*)(in + i);
        bf16x4 t;
        t[0] = (bf16)v.x; t[1] = (bf16)v.y; t[2] = (bf16)v.z; t[3] = (bf16)v.w;
        *(bf16x4*)(out + i) = t;
    } else {
        for (; i < n; ++i) out[i] = (bf16)in[i];
    }
}

// ---------------------------------------------------------------------------
// TDM: DMA one [tile_rows x 32] bf16 tile (row stride K elements) into LDS,
// padding 16 B after every 64 B so LDS rows land at stride LDSS=40 bf16.
// Rows beyond rows_avail read as zero (hardware OOB) -> ragged N for free.
// Descriptor layout per CDNA5 ISA 8.3/8.4 (D# group0/group1); this toolchain
// exposes the 6-arg builtin (g0, g1, g2, g3, extra, cpol).
// ---------------------------------------------------------------------------
__device__ inline void tdm_load_tile(unsigned lds_addr, const bf16* gsrc,
                                     int rows_avail, int tile_rows, int K)
{
    unsigned long long ga = (unsigned long long)(uintptr_t)gsrc;
    u32x4 g0;
    g0[0] = 1u;                                     // count=1, user descriptor
    g0[1] = lds_addr;                               // lds_addr (bytes)
    g0[2] = (unsigned)ga;                           // global_addr[31:0]
    g0[3] = ((unsigned)(ga >> 32) & 0x01ffffffu)    // global_addr[56:32]
          | (2u << 30);                             // type = 2 ("image")
    i32x8 g1;
    g1[0] = (1 << 16)                               // data_size = 2 bytes
          | (1 << 20)                               // pad_enable
          | (3 << 22)                               // pad_interval: 16 DWORDs
          | (3 << 25);                              // pad_amount: 4 DWORDs
    g1[1] = (K & 0xffff) << 16;                                   // td0 lo
    g1[2] = (((unsigned)K >> 16) & 0xffff)
          | ((rows_avail & 0xffff) << 16);                        // td0 hi|td1 lo
    g1[3] = (((unsigned)rows_avail >> 16) & 0xffff)
          | (TK << 16);                                           // td1 hi|tile_d0
    g1[4] = tile_rows & 0xffff;                                   // tile_dim1
    g1[5] = K;                                                    // d0_stride lo
    g1[6] = 0;
    g1[7] = 0;
    i32x4 z4; z4[0] = 0; z4[1] = 0; z4[2] = 0; z4[3] = 0;
    i32x8 z8;
    z8[0] = 0; z8[1] = 0; z8[2] = 0; z8[3] = 0;
    z8[4] = 0; z8[5] = 0; z8[6] = 0; z8[7] = 0;
    __builtin_amdgcn_tensor_load_to_lds(g0, g1, z4, z4, z8, 0);
}

// ---------------------------------------------------------------------------
// WMMA GEMM: C[M,N] = act( A[M,K](bf16) @ W[N,K](bf16)^T + bias [+Res] )
// mode: 1=bias, 2=relu, 4=residual. M % 256 == 0, K % 32 == 0, ragged N ok.
// ---------------------------------------------------------------------------
__device__ inline bf16x16 frag2(const bf16* p0, const bf16* p1)
{
    bf16x8 lo = *(const bf16x8*)p0;
    bf16x8 hi = *(const bf16x8*)p1;
    return __builtin_shufflevector(lo, hi, 0,1,2,3,4,5,6,7,8,9,10,11,12,13,14,15);
}

__global__ __launch_bounds__(256)
void gemm_bf16(const bf16* __restrict__ A, const bf16* __restrict__ W,
               const float* __restrict__ bias, const float* __restrict__ Res,
               float* __restrict__ C, int M, int N, int K, int mode)
{
    __shared__ __align__(16) bf16 As[2][TM * LDSS];   // 2 x 20 KB
    __shared__ __align__(16) bf16 Bs[2][TN * LDSS];   // 2 x 10 KB

    const int tid  = threadIdx.x;
    const int lane = tid & 31;
    const int wave = tid >> 5;
    const int wm   = wave & 3;      // 4 wave rows x 64
    const int wn   = wave >> 2;     // 2 wave cols x 64
    const int g    = lane >> 4;
    const int lr   = lane & 15;
    const int m0   = blockIdx.y * TM;
    const int n0   = blockIdx.x * TN;

    const unsigned ldsA[2] = { (unsigned)(uintptr_t)(void*)&As[0][0],
                               (unsigned)(uintptr_t)(void*)&As[1][0] };
    const unsigned ldsB[2] = { (unsigned)(uintptr_t)(void*)&Bs[0][0],
                               (unsigned)(uintptr_t)(void*)&Bs[1][0] };

    f32x8 acc[4][4];
#pragma unroll
    for (int mi = 0; mi < 4; ++mi)
#pragma unroll
        for (int ni = 0; ni < 4; ++ni)
#pragma unroll
            for (int r = 0; r < 8; ++r) acc[mi][ni][r] = 0.0f;

    // prologue: DMA tile 0 into buffer 0 (wave 0 drives the TDM)
    if (wave == 0) {
        tdm_load_tile(ldsA[0], A + (size_t)m0 * K, M - m0, TM, K);
        tdm_load_tile(ldsB[0], W + (size_t)n0 * K, N - n0, TN, K);
        __builtin_amdgcn_s_wait_tensorcnt(0);
    }
    __syncthreads();
    asm volatile("" ::: "memory");   // LDS written by TDM, invisible to alias analysis

    int cur = 0;
    for (int k0 = 0; k0 < K; k0 += TK) {
        const bool has_next = (k0 + TK) < K;
        if (has_next && wave == 0) {      // DMA next tile while we do WMMA
            tdm_load_tile(ldsA[cur ^ 1], A + (size_t)m0 * K + (k0 + TK), M - m0, TM, K);
            tdm_load_tile(ldsB[cur ^ 1], W + (size_t)n0 * K + (k0 + TK), N - n0, TN, K);
        }

        bf16x16 af[4], bfr[4];
#pragma unroll
        for (int mi = 0; mi < 4; ++mi) {
            const bf16* row = &As[cur][(wm * 64 + mi * 16 + lr) * LDSS];
            af[mi] = frag2(row + g * 8, row + 16 + g * 8);
        }
#pragma unroll
        for (int ni = 0; ni < 4; ++ni) {
            const bf16* col = &Bs[cur][(wn * 64 + ni * 16 + lr) * LDSS];
            bfr[ni] = frag2(col + g * 8, col + 16 + g * 8);
        }
#pragma unroll
        for (int mi = 0; mi < 4; ++mi)
#pragma unroll
            for (int ni = 0; ni < 4; ++ni)
                acc[mi][ni] = __builtin_amdgcn_wmma_f32_16x16x32_bf16(
                    false, af[mi], false, bfr[ni], (short)0, acc[mi][ni],
                    false, false);

        if (wave == 0)
            __builtin_amdgcn_s_wait_tensorcnt(0);
        __syncthreads();
        asm volatile("" ::: "memory");
        cur ^= 1;
    }

    // epilogue: C/D layout lane=col (lane&15), VGPR r -> row r + (lane>>4)*8
#pragma unroll
    for (int mi = 0; mi < 4; ++mi)
#pragma unroll
        for (int ni = 0; ni < 4; ++ni) {
            int col = n0 + wn * 64 + ni * 16 + lr;
            if (col >= N) continue;
            float bv = (mode & 1) ? bias[col] : 0.0f;
#pragma unroll
            for (int r = 0; r < 8; ++r) {
                int row = m0 + wm * 64 + mi * 16 + g * 8 + r;
                float v = acc[mi][ni][r] + bv;
                if (mode & 4) v += Res[(size_t)row * N + col];
                if (mode & 2) v = v > 0.0f ? v : 0.0f;
                C[(size_t)row * N + col] = v;
            }
        }
}

// ---------------------------------------------------------------------------
// im2col: images [B,3,160,160] -> [NTOK, 768], K index = c*256 + p*16 + q
// ---------------------------------------------------------------------------
__global__ __launch_bounds__(256)
void im2col(const float* __restrict__ im, float* __restrict__ out)
{
    int t = blockIdx.x;
    int b = t / L_, l = t % L_;
    int hp = l / 10, wp = l % 10;
    for (int k = threadIdx.x; k < D_; k += 256) {
        int c = k >> 8, p = (k >> 4) & 15, q = k & 15;
        out[(size_t)t * D_ + k] =
            im[(((size_t)b * 3 + c) * 160 + hp * 16 + p) * 160 + wp * 16 + q];
    }
}

// ---------------------------------------------------------------------------
// pe_gather: [NTOK, 2304] = [E_ds[id] | E_im[id] | E_pa[l]]
// ---------------------------------------------------------------------------
__global__ __launch_bounds__(256)
void pe_gather(const int* __restrict__ ds, const int* __restrict__ imi,
               const float* __restrict__ Eds, const float* __restrict__ Eim,
               const float* __restrict__ Epa, float* __restrict__ out)
{
    int t = blockIdx.x;
    int b = t / L_, l = t % L_;
    int di = ds[b * L_ + l];
    int ii = imi[b * L_ + l];
    float* o = out + (size_t)t * D3_;
    for (int k = threadIdx.x; k < D_; k += 256) {
        o[k]          = Eds[(size_t)di * D_ + k];
        o[D_ + k]     = Eim[(size_t)ii * D_ + k];
        o[2 * D_ + k] = Epa[(size_t)l * D_ + k];
    }
}

// ---------------------------------------------------------------------------
// layernorm per token (768 elems)
// ---------------------------------------------------------------------------
__global__ __launch_bounds__(256)
void layernorm(const float* __restrict__ in, const float* __restrict__ gam,
               const float* __restrict__ bet, float* __restrict__ out)
{
    __shared__ float red[256];
    int t = blockIdx.x;
    const float* x = in + (size_t)t * D_;
    float s = 0.0f;
    for (int i = threadIdx.x; i < D_; i += 256) s += x[i];
    red[threadIdx.x] = s; __syncthreads();
    for (int off = 128; off > 0; off >>= 1) {
        if (threadIdx.x < off) red[threadIdx.x] += red[threadIdx.x + off];
        __syncthreads();
    }
    float mean = red[0] / (float)D_; __syncthreads();
    float v = 0.0f;
    for (int i = threadIdx.x; i < D_; i += 256) {
        float d = x[i] - mean; v += d * d;
    }
    red[threadIdx.x] = v; __syncthreads();
    for (int off = 128; off > 0; off >>= 1) {
        if (threadIdx.x < off) red[threadIdx.x] += red[threadIdx.x + off];
        __syncthreads();
    }
    float inv = rsqrtf(red[0] / (float)D_ + 1e-5f);
    for (int i = threadIdx.x; i < D_; i += 256)
        out[(size_t)t * D_ + i] = (x[i] - mean) * inv * gam[i] + bet[i];
}

// ---------------------------------------------------------------------------
// attention per (head, batch): qkv [NTOK,2304] -> o [NTOK,768]
// ---------------------------------------------------------------------------
__global__ __launch_bounds__(128)
void attention(const float* __restrict__ qkv, float* __restrict__ o)
{
    const int h = blockIdx.x, b = blockIdx.y;
    __shared__ float Ks[L_ * DH_];
    __shared__ float Vs[L_ * DH_];
    __shared__ float qs[DH_];
    __shared__ float sc[128];
    __shared__ float red[128];
    const int tid = threadIdx.x;

    for (int i = tid; i < L_ * DH_; i += 128) {
        int l = i >> 6, d = i & 63;
        size_t base = ((size_t)(b * L_ + l)) * D3_ + h * DH_ + d;
        Ks[i] = qkv[base + D_];
        Vs[i] = qkv[base + 2 * D_];
    }
    __syncthreads();

    for (int q = 0; q < L_; ++q) {
        if (tid < DH_)
            qs[tid] = qkv[((size_t)(b * L_ + q)) * D3_ + h * DH_ + tid];
        __syncthreads();
        float s = -3.0e38f;
        if (tid < L_) {
            s = 0.0f;
            for (int d = 0; d < DH_; ++d) s += qs[d] * Ks[tid * DH_ + d];
            s *= 0.125f;
        }
        sc[tid] = s; red[tid] = s; __syncthreads();
        for (int off = 64; off > 0; off >>= 1) {
            if (tid < off) red[tid] = fmaxf(red[tid], red[tid + off]);
            __syncthreads();
        }
        float mx = red[0]; __syncthreads();
        float e = (tid < L_) ? expf(sc[tid] - mx) : 0.0f;
        sc[tid] = e; red[tid] = e; __syncthreads();
        for (int off = 64; off > 0; off >>= 1) {
            if (tid < off) red[tid] += red[tid + off];
            __syncthreads();
        }
        float sum = red[0]; __syncthreads();
        if (tid < DH_) {
            float a = 0.0f;
            for (int k = 0; k < L_; ++k) a += sc[k] * Vs[k * DH_ + tid];
            o[((size_t)(b * L_ + q)) * D_ + h * DH_ + tid] = a / sum;
        }
        __syncthreads();
    }
}

// ---------------------------------------------------------------------------
// gating: top-2 of 8 logits + softmax of the two
// ---------------------------------------------------------------------------
__global__ __launch_bounds__(128)
void gate_topk(const float* __restrict__ logits, float* __restrict__ topw,
               int* __restrict__ topi)
{
    int t = blockIdx.x * 128 + threadIdx.x;
    if (t >= NTOK) return;
    const float* gl = logits + (size_t)t * NE_;
    int i0 = 0; float v0 = gl[0];
    for (int i = 1; i < NE_; ++i) if (gl[i] > v0) { v0 = gl[i]; i0 = i; }
    int i1 = (i0 == 0) ? 1 : 0; float v1 = gl[i1];
    for (int i = 0; i < NE_; ++i)
        if (i != i0 && gl[i] > v1) { v1 = gl[i]; i1 = i; }
    float e1 = expf(v1 - v0);
    float inv = 1.0f / (1.0f + e1);
    topw[t * 2] = inv; topw[t * 2 + 1] = e1 * inv;
    topi[t * 2] = i0;  topi[t * 2 + 1] = i1;
}

__global__ __launch_bounds__(256)
void moe_acc(const float* __restrict__ eo, const float* __restrict__ topw,
             const int* __restrict__ topi, int e, float* __restrict__ moe)
{
    int t = blockIdx.x;
    float w = 0.0f;
    if (topi[t * 2] == e) w = topw[t * 2];
    else if (topi[t * 2 + 1] == e) w = topw[t * 2 + 1];
    for (int k = threadIdx.x; k < D_; k += 256) {
        float v = w * eo[(size_t)t * D_ + k];
        if (e == 0) moe[(size_t)t * D_ + k] = v;
        else        moe[(size_t)t * D_ + k] += v;
    }
}

// ---------------------------------------------------------------------------
// bilinear upsample 10x10 -> 160x160, half-pixel centers, clamped edges
// ---------------------------------------------------------------------------
__global__ __launch_bounds__(256)
void upsample(const float* __restrict__ s, float* __restrict__ out)
{
    int idx = blockIdx.x * 256 + threadIdx.x;
    if (idx >= B_ * 160 * 160) return;
    int x = idx % 160, y = (idx / 160) % 160, b = idx / (160 * 160);
    float sy = (y + 0.5f) / 16.0f - 0.5f;
    float sx = (x + 0.5f) / 16.0f - 0.5f;
    int y0f = (int)floorf(sy); float fy = sy - (float)y0f;
    int x0f = (int)floorf(sx); float fx = sx - (float)x0f;
    int y0 = min(max(y0f, 0), 9), y1 = min(max(y0f + 1, 0), 9);
    int x0 = min(max(x0f, 0), 9), x1 = min(max(x0f + 1, 0), 9);
    const float* sb = s + (size_t)b * L_;
    float v = (1.0f - fy) * ((1.0f - fx) * sb[y0 * 10 + x0] + fx * sb[y0 * 10 + x1])
            +         fy  * ((1.0f - fx) * sb[y1 * 10 + x0] + fx * sb[y1 * 10 + x1]);
    out[idx] = v;
}

// ---------------------------------------------------------------------------
// contrastive + load-balance losses
// ---------------------------------------------------------------------------
__global__ __launch_bounds__(256)
void rownorm(const float* __restrict__ x, float* __restrict__ nrm)
{
    __shared__ float red[256];
    int t = blockIdx.x;
    float s = 0.0f;
    for (int i = threadIdx.x; i < D_; i += 256) {
        float v = x[(size_t)t * D_ + i]; s += v * v;
    }
    red[threadIdx.x] = s; __syncthreads();
    for (int off = 128; off > 0; off >>= 1) {
        if (threadIdx.x < off) red[threadIdx.x] += red[threadIdx.x + off];
        __syncthreads();
    }
    if (threadIdx.x == 0) nrm[t] = sqrtf(red[0]);
}

__global__ __launch_bounds__(128)
void contrast(const float* __restrict__ x, const float* __restrict__ nrm,
              const int* __restrict__ ds, float* __restrict__ acc)
{
    __shared__ float sim[128];
    __shared__ float red[128];
    __shared__ int   ired[128];
    int q = blockIdx.x % L_, b = blockIdx.x / L_;
    int tid = threadIdx.x;
    int qds = ds[b * L_ + q];
    float s = -3.0e38f;
    int eq = 0;
    if (tid < L_) {
        const float* xq = x + ((size_t)(b * L_ + q)) * D_;
        const float* xm = x + ((size_t)(b * L_ + tid)) * D_;
        float d = 0.0f;
        for (int i = 0; i < D_; ++i) d += xq[i] * xm[i];
        s = d / (nrm[b * L_ + q] * nrm[b * L_ + tid]) * 10.0f;  // / TEMP
        eq = (ds[b * L_ + tid] == qds) ? 1 : 0;
    }
    sim[tid] = s;
    red[tid] = (float)eq;
    ired[tid] = eq ? tid : (1 << 30);
    __syncthreads();
    for (int off = 64; off > 0; off >>= 1) {
        if (tid < off) {
            red[tid] += red[tid + off];
            ired[tid] = min(ired[tid], ired[tid + off]);
        }
        __syncthreads();
    }
    int pos_cnt = (int)red[0], first_pos = ired[0];
    __syncthreads();
    red[tid] = sim[tid]; __syncthreads();
    for (int off = 64; off > 0; off >>= 1) {
        if (tid < off) red[tid] = fmaxf(red[tid], red[tid + off]);
        __syncthreads();
    }
    float mx = red[0]; __syncthreads();
    red[tid] = (tid < L_) ? expf(sim[tid] - mx) : 0.0f; __syncthreads();
    for (int off = 64; off > 0; off >>= 1) {
        if (tid < off) red[tid] += red[tid + off];
        __syncthreads();
    }
    if (tid == 0) {
        if (pos_cnt > 1 && pos_cnt < L_) {
            float per = logf(red[0]) + mx - sim[first_pos];
            atomicAdd(&acc[0], per);
            atomicAdd(&acc[1], 1.0f);
        }
    }
}

__global__ __launch_bounds__(128)
void lb_usage(const float* __restrict__ topw, float* __restrict__ acc)
{
    __shared__ float r0[128], r1[128];
    int b = blockIdx.x, tid = threadIdx.x;
    float a = 0.0f, c = 0.0f;
    if (tid < L_) {
        a = topw[(size_t)(b * L_ + tid) * 2];
        c = topw[(size_t)(b * L_ + tid) * 2 + 1];
    }
    r0[tid] = a; r1[tid] = c; __syncthreads();
    for (int off = 64; off > 0; off >>= 1) {
        if (tid < off) { r0[tid] += r0[tid + off]; r1[tid] += r1[tid + off]; }
        __syncthreads();
    }
    if (tid == 0) {
        float u0 = r0[0] / (float)L_, u1 = r1[0] / (float)L_;
        float d = u0 - u1;
        atomicAdd(&acc[2], 0.5f * d * d);
    }
}

__global__ void zero_acc(float* acc)
{
    if (threadIdx.x < 4) acc[threadIdx.x] = 0.0f;
}

__global__ void finalize(const float* __restrict__ acc, float* __restrict__ out2)
{
    if (threadIdx.x == 0) {
        float cnt = acc[1] < 1.0f ? 1.0f : acc[1];
        out2[0] = acc[0] / cnt;
        out2[1] = acc[2] / (float)B_;
    }
}

// ===========================================================================
extern "C" void kernel_launch(void* const* d_in, const int* in_sizes, int n_in,
                              void* d_out, int out_size, void* d_ws, size_t ws_size,
                              hipStream_t stream)
{
    (void)in_sizes; (void)n_in; (void)out_size; (void)ws_size;
    auto F = [&](int i) { return (const float*)d_in[i]; };
    const float* images = F(0);
    const int* ds_ids = (const int*)d_in[1];
    const int* im_ids = (const int*)d_in[2];
    const float *Wp = F(3), *bp = F(4), *E_ds = F(5), *E_im = F(6), *E_pa = F(7);
    const float *posW1 = F(8), *posb1 = F(9), *posW2 = F(10), *posb2 = F(11);
    const float *Wg = F(84), *bg = F(85);
    const float *We1 = F(86), *be1 = F(87), *We2 = F(88), *be2 = F(89);
    const float *We3 = F(90), *be3 = F(91);
    const float *Ws1 = F(92), *bs1 = F(93), *Ws2 = F(94), *bs2 = F(95);

    float* ws = (float*)d_ws;
    size_t off = 0;
    float* X    = ws + off; off += (size_t)NTOK * D_;
    float* TA   = ws + off; off += (size_t)NTOK * D_;
    float* TB   = ws + off; off += (size_t)NTOK * D_;
    float* MOE  = ws + off; off += (size_t)NTOK * D_;
    float* EO   = ws + off; off += (size_t)NTOK * D_;
    float* BIG  = ws + off; off += (size_t)NTOK * D4_;
    float* GATE = ws + off; off += (size_t)NTOK * NE_;
    float* TOPW = ws + off; off += (size_t)NTOK * 2;
    int*   TOPI = (int*)(ws + off); off += (size_t)NTOK * 2;
    float* SEGS = ws + off; off += NTOK;
    float* NRM  = ws + off; off += NTOK;
    float* ACC  = ws + off; off += 8;
    bf16*  AB16 = (bf16*)(ws + off); off += (size_t)NTOK * D4_ / 2;   // A scratch
    bf16*  XB16 = (bf16*)(ws + off); off += (size_t)NTOK * D_ / 2;    // x bf16
    bf16*  WB   = (bf16*)(ws + off);                                   // weights
    float* H1 = BIG;
    float* H2 = BIG + (size_t)NTOK * HID_;
    float* seg_out  = (float*)d_out;
    float* scal_out = seg_out + (size_t)B_ * 160 * 160;

    auto CVT = [&](const float* src, bf16* dst, size_t n) {
        f32_to_bf16<<<(unsigned)((n / 4 + 255) / 256), 256, 0, stream>>>(src, dst, (long)n);
    };
    size_t wo = 0;
    auto WCVT = [&](const float* w, size_t n) {
        bf16* p = WB + wo; wo += n; CVT(w, p, n); return (const bf16*)p;
    };

    // ---- pre-convert all GEMM weights to bf16 (recomputed per call)
    const bf16* Wp16    = WCVT(Wp,    (size_t)D_ * D_);
    const bf16* posW116 = WCVT(posW1, (size_t)D_ * D3_);
    const bf16* posW216 = WCVT(posW2, (size_t)D_ * D_);
    const bf16 *Wqkv16[6], *Wo16[6], *W116[6], *W216[6];
    for (int l = 0; l < 6; ++l) {
        int b0 = 12 + 12 * l;
        Wqkv16[l] = WCVT(F(b0),     (size_t)D3_ * D_);
        Wo16[l]   = WCVT(F(b0 + 2), (size_t)D_ * D_);
        W116[l]   = WCVT(F(b0 + 6), (size_t)D4_ * D_);
        W216[l]   = WCVT(F(b0 + 8), (size_t)D_ * D4_);
    }
    const bf16* Wg16  = WCVT(Wg,  (size_t)NE_ * D_);
    const bf16* We116 = WCVT(We1, (size_t)NE_ * HID_ * D_);
    const bf16* We216 = WCVT(We2, (size_t)NE_ * HID_ * HID_);
    const bf16* We316 = WCVT(We3, (size_t)NE_ * D_ * HID_);
    const bf16* Ws116 = WCVT(Ws1, (size_t)DHALF_ * D_);
    const bf16* Ws216 = WCVT(Ws2, (size_t)1 * DHALF_);

    auto RUN = [&](const bf16* A16, const bf16* W16, const float* bias,
                   const float* Res, float* Cq, int M, int N, int K, int mode) {
        gemm_bf16<<<dim3((N + TN - 1) / TN, (M + TM - 1) / TM), 256, 0, stream>>>(
            A16, W16, bias, Res, Cq, M, N, K, mode);
    };
    auto GEMM = [&](const float* Af, const bf16* W16, const float* bias,
                    const float* Res, float* Cq, int M, int N, int K, int mode) {
        CVT(Af, AB16, (size_t)M * K);
        RUN(AB16, W16, bias, Res, Cq, M, N, K, mode);
    };

    // ---- patch embed + 3D position MLP
    im2col<<<NTOK, 256, 0, stream>>>(images, TA);
    GEMM(TA, Wp16, bp, nullptr, TB, NTOK, D_, D_, 1);
    pe_gather<<<NTOK, 256, 0, stream>>>(ds_ids, im_ids, E_ds, E_im, E_pa, BIG);
    GEMM(BIG, posW116, posb1, nullptr, MOE, NTOK, D_, D3_, 1 | 2);
    GEMM(MOE, posW216, posb2, TB, X, NTOK, D_, D_, 1 | 4);

    // ---- 6 post-norm transformer layers
    for (int l = 0; l < 6; ++l) {
        int b0 = 12 + 12 * l;
        const float *bqkv = F(b0 + 1), *bo = F(b0 + 3), *g1 = F(b0 + 4),
                    *be1n = F(b0 + 5), *bb1 = F(b0 + 7), *bb2 = F(b0 + 9),
                    *g2 = F(b0 + 10), *be2n = F(b0 + 11);
        GEMM(X, Wqkv16[l], bqkv, nullptr, BIG, NTOK, D3_, D_, 1);
        attention<<<dim3(NH_, B_), 128, 0, stream>>>(BIG, TA);
        GEMM(TA, Wo16[l], bo, X, TB, NTOK, D_, D_, 1 | 4);
        layernorm<<<NTOK, 256, 0, stream>>>(TB, g1, be1n, X);
        GEMM(X, W116[l], bb1, nullptr, BIG, NTOK, D4_, D_, 1 | 2);
        GEMM(BIG, W216[l], bb2, X, TB, NTOK, D_, D4_, 1 | 4);
        layernorm<<<NTOK, 256, 0, stream>>>(TB, g2, be2n, X);
    }

    // ---- MoE (dense all-expert + top-2 combine); x converted once
    CVT(X, XB16, (size_t)NTOK * D_);
    RUN(XB16, Wg16, bg, nullptr, GATE, NTOK, NE_, D_, 1);
    gate_topk<<<(NTOK + 127) / 128, 128, 0, stream>>>(GATE, TOPW, TOPI);
    for (int e = 0; e < NE_; ++e) {
        RUN(XB16, We116 + (size_t)e * HID_ * D_, be1 + (size_t)e * HID_,
            nullptr, H1, NTOK, HID_, D_, 1 | 2);
        GEMM(H1, We216 + (size_t)e * HID_ * HID_, be2 + (size_t)e * HID_,
             nullptr, H2, NTOK, HID_, HID_, 1 | 2);
        GEMM(H2, We316 + (size_t)e * D_ * HID_, be3 + (size_t)e * D_,
             nullptr, EO, NTOK, D_, HID_, 1);
        moe_acc<<<NTOK, 256, 0, stream>>>(EO, TOPW, TOPI, e, MOE);
    }

    // ---- segmentation head + upsample
    GEMM(MOE, Ws116, bs1, nullptr, TB, NTOK, DHALF_, D_, 1 | 2);
    GEMM(TB, Ws216, bs2, nullptr, SEGS, NTOK, 1, DHALF_, 1);
    upsample<<<(B_ * 160 * 160 + 255) / 256, 256, 0, stream>>>(SEGS, seg_out);

    // ---- losses
    zero_acc<<<1, 32, 0, stream>>>(ACC);
    rownorm<<<NTOK, 256, 0, stream>>>(X, NRM);
    contrast<<<NTOK, 128, 0, stream>>>(X, NRM, ds_ids, ACC);
    lb_usage<<<B_, 128, 0, stream>>>(TOPW, ACC);
    finalize<<<1, 32, 0, stream>>>(ACC, scal_out);
}
